// TAGConv_3l_512h_w_k3_70274254897752
// MI455X (gfx1250) — matrix operand order repaired and checked by reading, not compiled, not run on hardware
//
#include <hip/hip_runtime.h>
#include <math.h>

typedef __attribute__((ext_vector_type(16))) _Float16 v16h;
typedef __attribute__((ext_vector_type(8)))  float    v8f;

static constexpr int N_NODES = 100000;
static constexpr int N_EDGES = 1600000;
static constexpr int N_CLASS = 26;
static constexpr int HIDDEN  = 128;
static constexpr int KHOPS   = 3;

// ---------------------------------------------------------------- utilities

__global__ void zero_f32(float* __restrict__ p, int n) {
  int i = blockIdx.x * blockDim.x + threadIdx.x;
  if (i < n) p[i] = 0.0f;
}

__global__ void deg_kernel(const int* __restrict__ col, const float* __restrict__ w,
                           float* __restrict__ deg, int E) {
  int e = blockIdx.x * blockDim.x + threadIdx.x;
  if (e < E) atomicAdd(&deg[col[e]], w[e]);
}

__global__ void dinv_kernel(float* __restrict__ deg, int n) {
  int i = blockIdx.x * blockDim.x + threadIdx.x;
  if (i < n) {
    float d = deg[i];
    deg[i] = (d > 0.0f) ? rsqrtf(fmaxf(d, 1e-12f)) : 0.0f;
  }
}

__global__ void norm_kernel(const int* __restrict__ row, const int* __restrict__ col,
                            const float* __restrict__ w, const float* __restrict__ dinv,
                            float* __restrict__ nrm, int E) {
  int e = blockIdx.x * blockDim.x + threadIdx.x;
  if (e < E) nrm[e] = dinv[row[e]] * w[e] * dinv[col[e]];
}

// Pad x (N x 26) -> x_pad (N x 32), zero cols 26..31.
__global__ void pad_x_kernel(const float* __restrict__ x, float* __restrict__ xp, int n) {
  int i = blockIdx.x * blockDim.x + threadIdx.x;
  if (i < n) {
    int node = i >> 5, c = i & 31;
    xp[i] = (c < N_CLASS) ? x[node * N_CLASS + c] : 0.0f;
  }
}

__global__ void pad_b_kernel(const float* __restrict__ b, float* __restrict__ bp,
                             int true_n, int pad_n) {
  int i = blockIdx.x * blockDim.x + threadIdx.x;
  if (i < pad_n) bp[i] = (i < true_n) ? b[i] : 0.0f;
}

// Pack W (HOPS x true_in x true_out, f32 row-major) into f16 B-fragment order:
// P[(((hop*KT + kb)*NT + nt)*32 + lane)*16 + i] = W[hop][kb*32 + (i&7) + 8*(lane>>4) + 16*(i>>3)][nt*16 + (lane&15)]
__global__ void pack_w_kernel(const float* __restrict__ W, _Float16* __restrict__ P,
                              int true_in, int true_out, int KT, int NT, int total) {
  int idx = blockIdx.x * blockDim.x + threadIdx.x;
  if (idx >= total) return;
  int i    = idx & 15;
  int lane = (idx >> 4) & 31;
  int t    = idx >> 9;                 // (hop*KT + kb)*NT + nt
  int nt   = t % NT;
  int kb   = (t / NT) % KT;
  int hop  = t / (NT * KT);
  int k = kb * 32 + (i & 7) + 8 * (lane >> 4) + 16 * (i >> 3);
  int n = nt * 16 + (lane & 15);
  float v = (k < true_in && n < true_out) ? W[((size_t)hop * true_in + k) * true_out + n]
                                          : 0.0f;
  P[idx] = (_Float16)v;
}

// ------------------------------------------------------- sparse propagation
// One wave32 per edge. out[col] += norm * h[row]. Feature tables (<=51MB) are
// L2-resident (192MB), so the random gather + atomic scatter runs at L2 rate.

template <int F>
__global__ void spmv_kernel(const int* __restrict__ row, const int* __restrict__ col,
                            const float* __restrict__ nrm,
                            const float* __restrict__ hin, float* __restrict__ hout,
                            int E) {
  int wid  = (int)((blockIdx.x * blockDim.x + threadIdx.x) >> 5);
  int lane = threadIdx.x & 31;
  if (wid >= E) return;
  int   r  = row[wid];
  int   c  = col[wid];
  float nv = nrm[wid];
  const float* __restrict__ src = hin  + (size_t)r * F;
  float* dst = hout + (size_t)c * F;
  if constexpr (F == 32) {
    atomicAdd(&dst[lane], nv * src[lane]);
  } else {
    static_assert(F == 128, "F must be 32 or 128");
    const float4 v = *(const float4*)(src + lane * 4);
    atomicAdd(&dst[lane * 4 + 0], nv * v.x);
    atomicAdd(&dst[lane * 4 + 1], nv * v.y);
    atomicAdd(&dst[lane * 4 + 2], nv * v.z);
    atomicAdd(&dst[lane * 4 + 3], nv * v.w);
  }
}

// ----------------------------------------------------------- WMMA GEMM+acc
// dst = [acc +] H[M x IN_DIM] @ Wpacked  (+bias when !ACC), optional fused ELU.
// One wave per 16-row M tile, NT=OUT_DIM/16 col tiles in registers, KT=IN_DIM/32
// fully unrolled -> NT*KT static v_wmma_f32_16x16x32_f16 per kernel, no masking.

template <int IN_DIM, int OUT_DIM, bool ACC, bool ELU>
__global__ void wmma_gemm(const float* __restrict__ H, const _Float16* __restrict__ P,
                          const float* __restrict__ bias, const float* accbuf,
                          float* dst, int dst_ld, int store_cols) {
  constexpr int NT = OUT_DIM / 16;
  constexpr int KT = IN_DIM / 32;
  const int wave  = threadIdx.x >> 5;
  const int mtile = blockIdx.x * (blockDim.x >> 5) + wave;
  if (mtile * 16 >= N_NODES) return;               // wave-uniform exit

  const int lane = threadIdx.x & 31;
  const int lh   = lane >> 4;
  const int ll   = lane & 15;
  const int m0   = mtile * 16;

  v8f c[NT];
  #pragma unroll
  for (int nt = 0; nt < NT; ++nt) {
    if (ACC) {
      const float* ap = accbuf + (size_t)(m0 + 8 * lh) * OUT_DIM + nt * 16 + ll;
      #pragma unroll
      for (int r = 0; r < 8; ++r) c[nt][r] = ap[r * OUT_DIM];
    } else {
      const float bv = bias[nt * 16 + ll];
      #pragma unroll
      for (int r = 0; r < 8; ++r) c[nt][r] = bv;
    }
  }

  const float* hrow = H + (size_t)(m0 + ll) * IN_DIM;
  const _Float16* pb = P + lane * 16;
  #pragma unroll
  for (int kb = 0; kb < KT; ++kb) {
    // A fragment 16x32 f16 (ISA 7.12.2): lane half lh covers K = {lh*8..lh*8+7, 16+lh*8..16+lh*8+7}
    const float* hp = hrow + kb * 32 + lh * 8;
    const float4 p0 = *(const float4*)(hp + 0);
    const float4 p1 = *(const float4*)(hp + 4);
    const float4 p2 = *(const float4*)(hp + 16);
    const float4 p3 = *(const float4*)(hp + 20);
    v16h a;
    a[0]  = (_Float16)p0.x; a[1]  = (_Float16)p0.y; a[2]  = (_Float16)p0.z; a[3]  = (_Float16)p0.w;
    a[4]  = (_Float16)p1.x; a[5]  = (_Float16)p1.y; a[6]  = (_Float16)p1.z; a[7]  = (_Float16)p1.w;
    a[8]  = (_Float16)p2.x; a[9]  = (_Float16)p2.y; a[10] = (_Float16)p2.z; a[11] = (_Float16)p2.w;
    a[12] = (_Float16)p3.x; a[13] = (_Float16)p3.y; a[14] = (_Float16)p3.z; a[15] = (_Float16)p3.w;
    #pragma unroll
    for (int nt = 0; nt < NT; ++nt) {
      const v16h b = *(const v16h*)(pb + (size_t)(kb * NT + nt) * 512);
      c[nt] = __builtin_amdgcn_wmma_f32_16x16x32_f16(
          false, a, false, b, (short)0, c[nt], false, false);
    }
  }

  #pragma unroll
  for (int nt = 0; nt < NT; ++nt) {
    const int n = nt * 16 + ll;
    if (n < store_cols) {
      float* dp = dst + (size_t)(m0 + 8 * lh) * dst_ld + n;
      #pragma unroll
      for (int r = 0; r < 8; ++r) {
        float v = c[nt][r];
        if (ELU) v = (v > 0.0f) ? v : (expf(v) - 1.0f);
        dp[r * dst_ld] = v;
      }
    }
  }
}

// ------------------------------------------------------------- host helpers

static inline int cdiv(int a, int b) { return (a + b - 1) / b; }

static void launch_zero(float* p, int n, hipStream_t s) {
  zero_f32<<<cdiv(n, 256), 256, 0, s>>>(p, n);
}

template <int IN, int OUT, bool ACC, bool ELU>
static void launch_gemm(const float* H, const _Float16* P, const float* bias,
                        const float* accbuf, float* dst, int dst_ld, int store_cols,
                        hipStream_t s) {
  const int MT = cdiv(N_NODES, 16);               // 6250 tiles
  wmma_gemm<IN, OUT, ACC, ELU><<<cdiv(MT, 4), 128, 0, s>>>(H, P, bias, accbuf, dst,
                                                           dst_ld, store_cols);
}

template <int F>
static void launch_spmv(const int* row, const int* col, const float* nrm,
                        const float* hin, float* hout, hipStream_t s) {
  spmv_kernel<F><<<cdiv(N_EDGES * 32, 256), 256, 0, s>>>(row, col, nrm, hin, hout, N_EDGES);
}

// One TAGConv layer: final = [elu]( sum_k (A^k X) @ W[k] + b ).
template <int IN, int OUT, bool FINAL_ELU>
static void tag_layer(const float* X, const _Float16* P, const float* bias,
                      float* acc, float* hA, float* hB,
                      float* final_dst, int final_ld, int final_cols,
                      const int* row, const int* col, const float* nrm, hipStream_t s) {
  constexpr int PSTRIDE = (IN / 32) * (OUT / 16) * 512;  // halves per hop
  launch_gemm<IN, OUT, false, false>(X, P, bias, acc, acc, OUT, OUT, s);
  const float* cur = X;
  float* nxt = hA;
  for (int k = 1; k <= KHOPS; ++k) {
    launch_zero(nxt, N_NODES * IN, s);
    launch_spmv<IN>(row, col, nrm, cur, nxt, s);
    const _Float16* Pk = P + (size_t)k * PSTRIDE;
    if (k < KHOPS)
      launch_gemm<IN, OUT, true, false>(nxt, Pk, bias, acc, acc, OUT, OUT, s);
    else
      launch_gemm<IN, OUT, true, FINAL_ELU>(nxt, Pk, bias, acc, final_dst, final_ld,
                                            final_cols, s);
    cur = nxt;
    nxt = (nxt == hA) ? hB : hA;
  }
}

// ------------------------------------------------------------------- entry

extern "C" void kernel_launch(void* const* d_in, const int* in_sizes, int n_in,
                              void* d_out, int out_size, void* d_ws, size_t ws_size,
                              hipStream_t stream) {
  const float* x  = (const float*)d_in[0];
  const int*   ei = (const int*)d_in[1];
  const float* ew = (const float*)d_in[2];
  const float* W1 = (const float*)d_in[3];
  const float* b1 = (const float*)d_in[4];
  const float* W2 = (const float*)d_in[5];
  const float* b2 = (const float*)d_in[6];
  const float* W3 = (const float*)d_in[7];
  const float* b3 = (const float*)d_in[8];
  float* out = (float*)d_out;

  const int* row = ei;               // edge_index[0]
  const int* col = ei + N_EDGES;     // edge_index[1]

  // workspace layout (float offsets, 32B-aligned blocks)
  float* ws   = (float*)d_ws;
  float* deg  = ws;                                  // 100096
  float* nrm  = ws + 100096;                         // 1600000
  float* xpad = ws + 1700096;                        // N*32
  float* h0   = ws + 4900096;                        // N*128
  float* hA   = ws + 17700096;                       // N*128
  float* hB   = ws + 30500096;                       // N*128
  float* acc  = ws + 43300096;                       // N*128
  float* b3p  = ws + 56100096;                       // 32
  _Float16* P1 = (_Float16*)(ws + 56100128);         // 4*1*8*512 = 16384 halves
  _Float16* P2 = P1 + 16384;                         // 4*4*8*512 = 65536 halves
  _Float16* P3 = P2 + 65536;                         // 4*4*2*512 = 16384 halves

  // ---- preconditioning: gcn_norm + padding + weight packing
  launch_zero(deg, N_NODES, stream);
  deg_kernel<<<cdiv(N_EDGES, 256), 256, 0, stream>>>(col, ew, deg, N_EDGES);
  dinv_kernel<<<cdiv(N_NODES, 256), 256, 0, stream>>>(deg, N_NODES);
  norm_kernel<<<cdiv(N_EDGES, 256), 256, 0, stream>>>(row, col, ew, deg, nrm, N_EDGES);

  pad_x_kernel<<<cdiv(N_NODES * 32, 256), 256, 0, stream>>>(x, xpad, N_NODES * 32);
  pad_b_kernel<<<1, 32, 0, stream>>>(b3, b3p, N_CLASS, 32);
  pack_w_kernel<<<cdiv(16384, 256), 256, 0, stream>>>(W1, P1, N_CLASS, HIDDEN, 1, 8, 16384);
  pack_w_kernel<<<cdiv(65536, 256), 256, 0, stream>>>(W2, P2, HIDDEN, HIDDEN, 4, 8, 65536);
  pack_w_kernel<<<cdiv(16384, 256), 256, 0, stream>>>(W3, P3, HIDDEN, N_CLASS, 4, 2, 16384);

  // ---- layer 1: 32(pad 26) -> 128, fused ELU into h0
  tag_layer<32, 128, true>(xpad, P1, b1, acc, hA, hB, h0, HIDDEN, HIDDEN,
                           row, col, nrm, stream);
  // ---- layer 2: 128 -> 128, fused ELU into h0
  tag_layer<128, 128, true>(h0, P2, b2, acc, hA, hB, h0, HIDDEN, HIDDEN,
                            row, col, nrm, stream);
  // ---- layer 3: 128 -> 32(pad 26), final store masked to d_out (ld=26)
  tag_layer<128, 32, false>(h0, P3, b3p, acc, hA, hB, out, N_CLASS, N_CLASS,
                            row, col, nrm, stream);
}
